// RGCN_86414741995956
// MI455X (gfx1250) — matrix operand order repaired and checked by reading, not compiled, not run on hardware
//
#include <hip/hip_runtime.h>

typedef __attribute__((ext_vector_type(16))) _Float16 v16h;
typedef __attribute__((ext_vector_type(8)))  float    v8f;

namespace {

constexpr int NN   = 100000;   // nodes
constexpr int EE   = 500000;   // edges per relation
constexpr int RR   = 3;        // relations
constexpr int DIN  = 128;
constexpr int DH   = 128;
constexpr int DOUT = 64;

__global__ void zero_kernel(float* __restrict__ p, int n) {
  int i = blockIdx.x * blockDim.x + threadIdx.x;
  if (i < n) p[i] = 0.0f;
}

// Degrees: one atomicAdd per endpoint, grid.y = relation.
__global__ void degree_kernel(const int* __restrict__ src, const int* __restrict__ dst,
                              float* __restrict__ degOut, float* __restrict__ degIn) {
  int e = blockIdx.x * blockDim.x + threadIdx.x;
  int r = blockIdx.y;
  if (e < EE) {
    atomicAdd(&degOut[r * NN + src[(size_t)r * EE + e]], 1.0f);
    atomicAdd(&degIn [r * NN + dst[(size_t)r * EE + e]], 1.0f);
  }
}

// In-place: deg -> rsqrt(max(deg,1))
__global__ void rsqrt_kernel(float* __restrict__ p, int n) {
  int i = blockIdx.x * blockDim.x + threadIdx.x;
  if (i < n) p[i] = rsqrtf(fmaxf(p[i], 1.0f));
}

// Pre-convert W[r] (fp32, [128 x D] row-major) into this file's contiguous-k WMMA-B
// layout as packed f16 tiles: Bh[r][(colTile*4+kk)*32 + lane][0..15], where
// element i = W[kk*32 + 16*(lane>>4) + i][colTile*16 + (lane&15)].
// One thread per (colTile,kk,lane) triple; writes 32 contiguous bytes.
template<int COLT>   // COLT = D/16
__global__ void convert_w_kernel(const float* __restrict__ W, _Float16* __restrict__ Bh) {
  constexpr int D   = COLT * 16;
  constexpr int TOT = COLT * 4 * 32;
  int t = blockIdx.x * blockDim.x + threadIdx.x;
  int r = blockIdx.y;
  if (t >= TOT) return;
  int lane = t & 31, kk = (t >> 5) & 3, colTile = t >> 7;
  int half = lane >> 4, sub = lane & 15;
  const float* wsrc = W + (size_t)r * 128 * D
                        + (size_t)(kk * 32 + 16 * half) * D + colTile * 16 + sub;
  v16h b;
  #pragma unroll
  for (int i = 0; i < 16; ++i) b[i] = (_Float16)wsrc[(size_t)i * D];
  *(v16h*)(Bh + ((size_t)r * TOT + t) * 16) = b;
}

// Persistent-B GEMM: C[NN x D] = A[NN x 128] * W[128 x D].
// Each wave owns one column tile; its entire B operand (4 k-chunks, 32 VGPRs
// of f16) is loaded once and held in registers while it strides over row
// tiles. Per row tile: 16x b128 A loads + v_cvt_pk conversions + 4 WMMAs.
template<int COLT, int SPC>   // COLT = D/16, SPC = row strips per column
__global__ void gemm_wmma_kernel(const float* __restrict__ A, const _Float16* __restrict__ Bh,
                                 float* __restrict__ C) {
  constexpr int D    = COLT * 16;
  constexpr int ROWT = NN / 16;
  const int lane = threadIdx.x & 31;
  const int w = blockIdx.x * (blockDim.x >> 5) + (threadIdx.x >> 5);
  if (w >= COLT * SPC) return;               // wave-uniform
  const int colTile = w % COLT;
  const int rt0     = w / COLT;
  const int half = lane >> 4, sub = lane & 15;

  v16h b[4];
  #pragma unroll
  for (int kk = 0; kk < 4; ++kk)
    b[kk] = *(const v16h*)(Bh + ((size_t)((colTile * 4 + kk) * 32 + lane)) * 16);

  for (int rt = rt0; rt < ROWT; rt += SPC) {
    const float* ap = A + (size_t)(rt * 16 + sub) * 128 + 16 * half;
    v8f c = {};
    #pragma unroll
    for (int kk = 0; kk < 4; ++kk) {
      const float* p = ap + kk * 32;
      float4 q0 = *(const float4*)(p);        // K = 16*half + 0..3
      float4 q1 = *(const float4*)(p + 4);    // K = 16*half + 4..7
      float4 q2 = *(const float4*)(p + 8);    // K = 16*half + 8..11
      float4 q3 = *(const float4*)(p + 12);   // K = 16*half + 12..15
      v16h a;
      a[0]  = (_Float16)q0.x; a[1]  = (_Float16)q0.y;
      a[2]  = (_Float16)q0.z; a[3]  = (_Float16)q0.w;
      a[4]  = (_Float16)q1.x; a[5]  = (_Float16)q1.y;
      a[6]  = (_Float16)q1.z; a[7]  = (_Float16)q1.w;
      a[8]  = (_Float16)q2.x; a[9]  = (_Float16)q2.y;
      a[10] = (_Float16)q2.z; a[11] = (_Float16)q2.w;
      a[12] = (_Float16)q3.x; a[13] = (_Float16)q3.y;
      a[14] = (_Float16)q3.z; a[15] = (_Float16)q3.w;
      // (neg_a, A, neg_b, B, c_mod, C, reuse_a, reuse_b)
      c = __builtin_amdgcn_wmma_f32_16x16x32_f16(false, a, false, b[kk], (short)0, c,
                                                 false, false);
    }
    float* cp = C + (size_t)(rt * 16 + 8 * half) * D + colTile * 16 + sub;
    #pragma unroll
    for (int v = 0; v < 8; ++v) cp[(size_t)v * D] = c[v];
  }
}

// One wave32 per edge: acc[dst] += sOut[src]*sIn[dst] * G[src]  (D floats).
// Lane owns D/32 contiguous floats -> one b128/b64 load + per-float atomics.
template<int D>
__global__ void scatter_kernel(const float* __restrict__ G, const int* __restrict__ src,
                               const int* __restrict__ dst, const float* __restrict__ sOut,
                               const float* __restrict__ sIn, float* __restrict__ acc) {
  const int e = blockIdx.x * (blockDim.x >> 5) + (threadIdx.x >> 5);
  if (e >= EE) return;
  const int lane = threadIdx.x & 31;
  const int s = src[e];
  const int d = dst[e];
  const float scale = sOut[s] * sIn[d];
  if constexpr (D == 128) {
    const float4 v = *(const float4*)(G + (size_t)s * D + lane * 4);
    float* a = acc + (size_t)d * D + lane * 4;
    atomicAdd(a + 0, scale * v.x);
    atomicAdd(a + 1, scale * v.y);
    atomicAdd(a + 2, scale * v.z);
    atomicAdd(a + 3, scale * v.w);
  } else {
    const float2 v = *(const float2*)(G + (size_t)s * D + lane * 2);
    float* a = acc + (size_t)d * D + lane * 2;
    atomicAdd(a + 0, scale * v.x);
    atomicAdd(a + 1, scale * v.y);
  }
}

// h = relu(h + sum_r b1[r])
__global__ void bias_relu_kernel(float* __restrict__ h, const float* __restrict__ b1) {
  int idx = blockIdx.x * blockDim.x + threadIdx.x;
  if (idx < NN * DH) {
    int dcol = idx & (DH - 1);
    float v = h[idx] + b1[dcol] + b1[DH + dcol] + b1[2 * DH + dcol];
    h[idx] = v > 0.0f ? v : 0.0f;
  }
}

// out += sum_r b2[r]
__global__ void bias_out_kernel(float* __restrict__ out, const float* __restrict__ b2) {
  int idx = blockIdx.x * blockDim.x + threadIdx.x;
  if (idx < NN * DOUT) {
    int dcol = idx & (DOUT - 1);
    out[idx] += b2[dcol] + b2[DOUT + dcol] + b2[2 * DOUT + dcol];
  }
}

inline int nblk(long n, int b) { return (int)((n + b - 1) / b); }

} // namespace

extern "C" void kernel_launch(void* const* d_in, const int* in_sizes, int n_in,
                              void* d_out, int out_size, void* d_ws, size_t ws_size,
                              hipStream_t stream) {
  (void)in_sizes; (void)n_in; (void)out_size; (void)ws_size;
  const float* x  = (const float*)d_in[0];   // [N, DIN]
  const float* W1 = (const float*)d_in[1];   // [R, DIN, DH]
  const float* b1 = (const float*)d_in[2];   // [R, DH]
  const float* W2 = (const float*)d_in[3];   // [R, DH, DOUT]
  const float* b2 = (const float*)d_in[4];   // [R, DOUT]
  const int*  src = (const int*)d_in[5];     // [R, E]
  const int*  dst = (const int*)d_in[6];     // [R, E]
  float* out = (float*)d_out;                // [N, DOUT]

  // workspace: sOut[R*N] | sIn[R*N] | h[N*DH] | G[N*DH] | Bh1 | Bh2 (~105 MB)
  float* wsf  = (float*)d_ws;
  float* sOut = wsf;
  float* sIn  = wsf + (size_t)RR * NN;
  float* h    = sIn + (size_t)RR * NN;
  float* G    = h   + (size_t)NN * DH;
  _Float16* Bh1 = (_Float16*)(G + (size_t)NN * DH);     // R * 8*128*16 halves
  _Float16* Bh2 = Bh1 + (size_t)RR * (DH / 16) * 128 * 16;

  const int B = 256;            // 8 wave32s per block
  constexpr int SPC = 1024;     // row strips per column tile in the GEMM

  // init + weight pre-conversion (independent of degree pass)
  zero_kernel<<<nblk((long)2 * RR * NN, B), B, 0, stream>>>(sOut, 2 * RR * NN);
  zero_kernel<<<nblk((long)NN * DH, B), B, 0, stream>>>(h, NN * DH);
  zero_kernel<<<nblk((long)NN * DOUT, B), B, 0, stream>>>(out, NN * DOUT);
  convert_w_kernel<DH / 16>
      <<<dim3(nblk((DH / 16) * 128, B), RR), B, 0, stream>>>(W1, Bh1);
  convert_w_kernel<DOUT / 16>
      <<<dim3(nblk((DOUT / 16) * 128, B), RR), B, 0, stream>>>(W2, Bh2);

  // degrees (shared by both layers: same graphs) -> rsqrt(clip(deg,1))
  degree_kernel<<<dim3(nblk(EE, B), RR), B, 0, stream>>>(src, dst, sOut, sIn);
  rsqrt_kernel<<<nblk((long)2 * RR * NN, B), B, 0, stream>>>(sOut, 2 * RR * NN);

  // layer 1: per relation persistent-B WMMA GEMM, then fused-scale scatter
  for (int r = 0; r < RR; ++r) {
    gemm_wmma_kernel<DH / 16, SPC>
        <<<(DH / 16) * SPC / 8, B, 0, stream>>>(
            x, Bh1 + (size_t)r * (DH / 16) * 128 * 16, G);
    scatter_kernel<DH><<<nblk(EE, 8), B, 0, stream>>>(
        G, src + (size_t)r * EE, dst + (size_t)r * EE,
        sOut + (size_t)r * NN, sIn + (size_t)r * NN, h);
  }
  bias_relu_kernel<<<nblk((long)NN * DH, B), B, 0, stream>>>(h, b1);

  // layer 2
  for (int r = 0; r < RR; ++r) {
    gemm_wmma_kernel<DOUT / 16, SPC>
        <<<(DOUT / 16) * SPC / 8, B, 0, stream>>>(
            h, Bh2 + (size_t)r * (DOUT / 16) * 128 * 16, G);
    scatter_kernel<DOUT><<<nblk(EE, 8), B, 0, stream>>>(
        G, src + (size_t)r * EE, dst + (size_t)r * EE,
        sOut + (size_t)r * NN, sIn + (size_t)r * NN, out);
  }
  bias_out_kernel<<<nblk((long)NN * DOUT, B), B, 0, stream>>>(out, b2);
}
